// MinkEncoderDecoderLite_40699110096957
// MI455X (gfx1250) — matrix operand order repaired and checked by reading, not compiled, not run on hardware
//
#include <hip/hip_runtime.h>

typedef _Float16 f16;
typedef long long i64;
typedef _Float16 v16h __attribute__((ext_vector_type(16)));
typedef float v8f __attribute__((ext_vector_type(8)));

union AV {
  v16h v;
  f16 h[16];
  unsigned int u[8];
  uint4 q[2];
};

static __device__ __forceinline__ uint4 zero4() {
  uint4 z; z.x = 0; z.y = 0; z.z = 0; z.w = 0; return z;
}

static __device__ __forceinline__ v8f wmma_f16(v16h a, v16h b, v8f c) {
  // D = A(16x32,f16) * B(32x16,f16) + C(16x16,f32)
  return __builtin_amdgcn_wmma_f32_16x16x32_f16(false, a, false, b, (short)0, c, false, false);
}

// ---------------------------------------------------------------------------
// Weight fragment packing: fp32 weights -> exact per-lane WMMA B-fragment
// layout; conv inner loops then load B with two b128 loads (L1/L2 resident).
// Fragment layout: PB[kstep][lane][k] (k=0..15), 32B contiguous per lane.
// B lane layout (32x16): lane<16 -> K{0..7,16..23} of col N=lane, lane>=16 -> K{8..15,24..31}
// ---------------------------------------------------------------------------

__global__ void k_packw_pair(const float* __restrict__ w, f16* __restrict__ pb,
                             int T, int KB) {
  int i = blockIdx.x * 256 + threadIdx.x;
  if (i >= KB * 32 * 16) return;
  int k = i & 15, lane = (i >> 4) & 31, kb = i >> 9;
  int half = lane >> 4, mrow = lane & 15;
  int tb = kb * 2 + (k >> 3);
  pb[i] = (tb < T) ? (f16)w[(tb * 16 + 8 * half + (k & 7)) * 16 + mrow] : (f16)0.f;
}

__global__ void k_packw32(const float* __restrict__ w, f16* __restrict__ pb) {
  int i = blockIdx.x * 256 + threadIdx.x;
  if (i >= 27 * 32 * 16) return;
  int k = i & 15, lane = (i >> 4) & 31, tap = i >> 9;
  int half = lane >> 4, mrow = lane & 15;
  pb[i] = (f16)w[(tap * 32 + 16 * (k >> 3) + 8 * half + (k & 7)) * 16 + mrow];
}

__global__ void k_packw_up(const float* __restrict__ w, f16* __restrict__ pb) {
  int i = blockIdx.x * 256 + threadIdx.x;
  if (i >= 8 * 32 * 16) return;
  int k = i & 15, lane = (i >> 4) & 31, o = i >> 9;
  int half = lane >> 4, mrow = lane & 15;
  pb[i] = (k < 8) ? (f16)w[(o * 16 + 8 * half + k) * 16 + mrow] : (f16)0.f;
}

__global__ void k_packw_d(const float* __restrict__ w, f16* __restrict__ pb) {
  int i = blockIdx.x * 256 + threadIdx.x;
  if (i >= 32 * 16) return;
  int k = i & 15, lane = i >> 4;
  int half = lane >> 4, mrow = lane & 15;
  pb[i] = (f16)w[(16 * (k >> 3) + 8 * half + (k & 7)) * 16 + mrow];
}

// ---------------------------------------------------------------------------
// Elementwise / reduction kernels (bandwidth-bound thin passes)
// ---------------------------------------------------------------------------

__global__ void k_zero(float* p, int n) {
  int i = blockIdx.x * blockDim.x + threadIdx.x;
  if (i < n) p[i] = 0.f;
}

__global__ void k_pool(const float* __restrict__ mi, float* __restrict__ mo,
                       int D, int H, int W) { // out dims
  i64 i = (i64)blockIdx.x * 256 + threadIdx.x;
  i64 n = (i64)D * H * W;
  if (i >= n) return;
  int w = (int)(i % W); i64 t = i / W; int h = (int)(t % H); int d = (int)(t / H);
  int Wi = 2 * W, Hi = 2 * H;
  float mx = 0.f;
  for (int oz = 0; oz < 2; ++oz)
    for (int oy = 0; oy < 2; ++oy)
      for (int ox = 0; ox < 2; ++ox)
        mx = fmaxf(mx, mi[(((i64)(2 * d + oz) * Hi) + (2 * h + oy)) * Wi + (2 * w + ox)]);
  mo[i] = mx;
}

__global__ void k_mulmask(const float* __restrict__ x, const float* __restrict__ m,
                          f16* __restrict__ o, i64 nvx) {
  i64 i = (i64)blockIdx.x * 256 + threadIdx.x;
  if (i >= nvx * 16) return;
  o[i] = (f16)(x[i] * m[i >> 4]);
}

__global__ void k_cat(const f16* __restrict__ a, const f16* __restrict__ b,
                      f16* __restrict__ o, i64 nvx) {
  i64 i = (i64)blockIdx.x * 256 + threadIdx.x;
  if (i >= nvx * 32) return;
  i64 v = i >> 5; int c = (int)(i & 31);
  o[i] = (c < 16) ? a[v * 16 + c] : b[v * 16 + (c - 16)];
}

// masked BN stats: stats[c]=sum(x*m), stats[C+c]=sum(x^2*m), stats[2C]=sum(m)
__global__ __launch_bounds__(256)
void k_bnstats(const f16* __restrict__ x, const float* __restrict__ m,
               float* __restrict__ stats, i64 nvx, int C) {
  __shared__ float sdata[256];
  int tid = threadIdx.x;
  int c = tid % C, slot = tid / C, vpb = 256 / C;
  float sx = 0.f, sx2 = 0.f, sn = 0.f;
  for (i64 v = (i64)blockIdx.x * vpb + slot; v < nvx; v += (i64)gridDim.x * vpb) {
    float mm = m[v];
    float xv = (float)x[v * (i64)C + c];
    float xm = xv * mm;
    sx += xm; sx2 += xm * xv;
    if (c == 0) sn += mm;
  }
  sdata[tid] = sx; __syncthreads();
  if (slot == 0) { float t = 0.f; for (int s = 0; s < vpb; ++s) t += sdata[s * C + c]; atomicAdd(&stats[c], t); }
  __syncthreads();
  sdata[tid] = sx2; __syncthreads();
  if (slot == 0) { float t = 0.f; for (int s = 0; s < vpb; ++s) t += sdata[s * C + c]; atomicAdd(&stats[C + c], t); }
  __syncthreads();
  sdata[tid] = sn; __syncthreads();
  if (tid == 0) { float t = 0.f; for (int s = 0; s < vpb; ++s) t += sdata[s * C]; atomicAdd(&stats[2 * C], t); }
}

__global__ void k_bnfinal(const float* __restrict__ stats, const float* __restrict__ g,
                          const float* __restrict__ b, float* __restrict__ sb, int C) {
  int c = threadIdx.x;
  if (c >= C) return;
  float n = stats[2 * C];
  float mean = stats[c] / n;
  float var = stats[C + c] / n - mean * mean;
  float rs = rsqrtf(var + 1e-5f);
  float sc = g[c] * rs;
  sb[c] = sc;
  sb[C + c] = b[c] - mean * sc;
}

// out = act((x*scale+bias)*m) ; act: 1=relu 2=lrelu(0.01)
__global__ void k_bnapply(const f16* __restrict__ x, const float* __restrict__ m,
                          const float* __restrict__ sb, f16* __restrict__ o,
                          i64 nvx, int C, int act) {
  i64 i = (i64)blockIdx.x * 256 + threadIdx.x;
  if (i >= nvx * (i64)C) return;
  i64 v = i / C; int c = (int)(i % C);
  float y = (float)x[i] * sb[c] + sb[C + c];
  y *= m[v];
  if (act == 1) y = fmaxf(y, 0.f);
  else if (act == 2) y = (y > 0.f) ? y : 0.01f * y;
  o[i] = (f16)y;
}

__global__ void k_resout(const f16* __restrict__ x, const f16* __restrict__ cv,
                         const float* __restrict__ m, f16* __restrict__ o, i64 nvx) {
  i64 i = (i64)blockIdx.x * 256 + threadIdx.x;
  if (i >= nvx * 16) return;
  float y = (float)x[i] + (float)cv[i];
  o[i] = (f16)(fmaxf(y, 0.f) * m[i >> 4]);
}

__global__ void k_resout_f32(const f16* __restrict__ x, const f16* __restrict__ cv,
                             const float* __restrict__ m, float* __restrict__ o, i64 nvx) {
  i64 i = (i64)blockIdx.x * 256 + threadIdx.x;
  if (i >= nvx * 16) return;
  float y = (float)x[i] + (float)cv[i];
  o[i] = fmaxf(y, 0.f) * m[i >> 4];
}

// ---------------------------------------------------------------------------
// WMMA implicit-GEMM conv kernels, templated on level geometry so every tap
// offset is a compile-time constant (folds into global_load IOFFSET; %,/ on
// power-of-two dims become masks/shifts).
// Wave tile: M = 16 consecutive voxels, N = 16 output channels.
// A lane layout (16-bit A 16x32): lane<16 -> K{0..7,16..23}, lane>=16 -> K{8..15,24..31}
// C/D layout: row M = r + 8*(lane/16), col N = lane%16.
// ---------------------------------------------------------------------------

template <int CIN, int D, int H, int W>
__global__ __launch_bounds__(256)
void k_conv3(const f16* __restrict__ in, const f16* __restrict__ pb,
             f16* __restrict__ outp) {
  const int lane = threadIdx.x & 31, wave = threadIdx.x >> 5;
  const int half = lane >> 4, mrow = lane & 15;
  const i64 base = ((i64)blockIdx.x * 8 + wave) * 16;
  const i64 vrow = base + mrow;
  const int wv = (int)(vrow % W);
  const int hv = (int)((vrow / W) % H);
  const int dv = (int)(vrow / ((i64)W * H));
  const f16* cen = in + vrow * CIN;   // center voxel; tap offsets are constants

  v8f acc = {};
  if (CIN == 16) {
    #pragma unroll
    for (int kb = 0; kb < 14; ++kb) {   // 27 taps + zero pad, 2 taps per WMMA
      AV a, b;
      #pragma unroll
      for (int s = 0; s < 2; ++s) {
        const int tap = kb * 2 + s;
        bool ok = false;
        if (tap < 27) {
          const int dz = tap / 9 - 1, dy = (tap / 3) % 3 - 1, dx = tap % 3 - 1;
          const int toff = ((dz * H + dy) * W + dx) * 16;   // compile-time
          if ((unsigned)(dv + dz) < (unsigned)D && (unsigned)(hv + dy) < (unsigned)H &&
              (unsigned)(wv + dx) < (unsigned)W) {
            a.q[s] = *(const uint4*)(cen + toff + 8 * half);
            ok = true;
          }
        }
        if (!ok) a.q[s] = zero4();
      }
      const uint4* qb = (const uint4*)(pb + ((kb * 32 + lane) << 4));
      b.q[0] = qb[0]; b.q[1] = qb[1];
      acc = wmma_f16(a.v, b.v, acc);
    }
  } else { // CIN == 32: one tap per WMMA (K = 32 channels)
    #pragma unroll
    for (int tap = 0; tap < 27; ++tap) {
      AV a, b;
      const int dz = tap / 9 - 1, dy = (tap / 3) % 3 - 1, dx = tap % 3 - 1;
      const int toff = ((dz * H + dy) * W + dx) * 32;       // compile-time
      if ((unsigned)(dv + dz) < (unsigned)D && (unsigned)(hv + dy) < (unsigned)H &&
          (unsigned)(wv + dx) < (unsigned)W) {
        const uint4* p4 = (const uint4*)(cen + toff);
        a.q[0] = p4[half];       // chans 8*half..+7     (K 0..7 / 8..15)
        a.q[1] = p4[2 + half];   // chans 16+8*half..+7  (K 16..23 / 24..31)
      } else {
        a.q[0] = zero4(); a.q[1] = zero4();
      }
      const uint4* qb = (const uint4*)(pb + ((tap * 32 + lane) << 4));
      b.q[0] = qb[0]; b.q[1] = qb[1];
      acc = wmma_f16(a.v, b.v, acc);
    }
  }
  #pragma unroll
  for (int r = 0; r < 8; ++r) outp[(base + r + 8 * half) * 16 + mrow] = (f16)acc[r];
}

// stride-2 2x2x2 conv: 8 taps -> 4 WMMAs per 16-voxel output tile
template <int D, int H, int W>   // out dims
__global__ __launch_bounds__(256)
void k_down2(const f16* __restrict__ in, const f16* __restrict__ pb,
             f16* __restrict__ outp) {
  const int lane = threadIdx.x & 31, wave = threadIdx.x >> 5;
  const int half = lane >> 4, mrow = lane & 15;
  const i64 base = ((i64)blockIdx.x * 8 + wave) * 16;
  const i64 vrow = base + mrow;
  const int wv = (int)(vrow % W);
  const int hv = (int)((vrow / W) % H);
  const int dv = (int)(vrow / ((i64)W * H));
  const int Hi = 2 * H, Wi = 2 * W;
  const f16* cen = in + (((i64)(2 * dv) * Hi + (2 * hv)) * Wi + (2 * wv)) * 16;

  v8f acc = {};
  #pragma unroll
  for (int kb = 0; kb < 4; ++kb) {
    AV a, b;
    #pragma unroll
    for (int s = 0; s < 2; ++s) {
      const int tap = kb * 2 + s;
      const int oz = tap >> 2, oy = (tap >> 1) & 1, ox = tap & 1;
      const int toff = ((oz * Hi + oy) * Wi + ox) * 16;     // compile-time
      a.q[s] = *(const uint4*)(cen + toff + 8 * half);
    }
    const uint4* qb = (const uint4*)(pb + ((kb * 32 + lane) << 4));
    b.q[0] = qb[0]; b.q[1] = qb[1];
    acc = wmma_f16(a.v, b.v, acc);
  }
  #pragma unroll
  for (int r = 0; r < 8; ++r) outp[(base + r + 8 * half) * 16 + mrow] = (f16)acc[r];
}

// transpose conv ks=2 s=2: each fine voxel has exactly one parent tap.
// block = 16 coarse voxels; wave o (0..7) = child offset; 1 WMMA per wave.
template <int Dc, int Hc, int Wc>
__global__ __launch_bounds__(256)
void k_up2(const f16* __restrict__ in, const f16* __restrict__ pb,
           f16* __restrict__ outp) {
  const int lane = threadIdx.x & 31, o = threadIdx.x >> 5;
  const int half = lane >> 4, mrow = lane & 15;
  const i64 base = (i64)blockIdx.x * 16;
  const i64 vrow = base + mrow;

  AV a, b;
  a.q[0] = *(const uint4*)(in + vrow * 16 + 8 * half);
  a.q[1] = zero4();                       // K 16..31 zero-pad
  const uint4* qb = (const uint4*)(pb + ((o * 32 + lane) << 4));
  b.q[0] = qb[0]; b.q[1] = qb[1];
  v8f acc = {};
  acc = wmma_f16(a.v, b.v, acc);

  const int oz = o >> 2, oy = (o >> 1) & 1, ox = o & 1;
  const int Hf = 2 * Hc, Wf = 2 * Wc;
  #pragma unroll
  for (int r = 0; r < 8; ++r) {
    i64 vv = base + r + 8 * half;
    int wc = (int)(vv % Wc);
    int hc = (int)((vv / Wc) % Hc);
    int dc = (int)(vv / ((i64)Wc * Hc));
    i64 fidx = (((i64)(2 * dc + oz) * Hf) + (2 * hc + oy)) * Wf + (2 * wc + ox);
    outp[fidx * 16 + mrow] = (f16)acc[r];
  }
}

// 1x1 dense 32->16 (residual downsample wd): one WMMA per 16-voxel tile
__global__ __launch_bounds__(256)
void k_dense1x1(const f16* __restrict__ in, const f16* __restrict__ pb,
                f16* __restrict__ outp) {
  const int lane = threadIdx.x & 31, wave = threadIdx.x >> 5;
  const int half = lane >> 4, mrow = lane & 15;
  const i64 base = ((i64)blockIdx.x * 8 + wave) * 16;
  const i64 vrow = base + mrow;

  AV a, b;
  const uint4* p4 = (const uint4*)(in + vrow * 32);
  a.q[0] = p4[half];
  a.q[1] = p4[2 + half];
  const uint4* qb = (const uint4*)(pb + (lane << 4));
  b.q[0] = qb[0]; b.q[1] = qb[1];
  v8f acc = {};
  acc = wmma_f16(a.v, b.v, acc);
  #pragma unroll
  for (int r = 0; r < 8; ++r) outp[(base + r + 8 * half) * 16 + mrow] = (f16)acc[r];
}

// ---------------------------------------------------------------------------
// Host orchestration
// Param flattening (JAX pytree, dict keys sorted alphabetically):
//   [0]=x [1]=mask, then down1..4{b,g,w}, s{1..4}r{1,2}{ba,bb,ga,gb,w1,w2},
//   stem{b1,b2,g1,g2,w1,w2}, u{1..4}r1{ba,bb,ga,gb,w1,w2,wd}+u{i}r2{..no wd},
//   up1..4{b,g,w}  -> 132 leaves total
// ---------------------------------------------------------------------------

extern "C" void kernel_launch(void* const* d_in, const int* in_sizes, int n_in,
                              void* d_out, int out_size, void* d_ws, size_t ws_size,
                              hipStream_t stream) {
  (void)in_sizes; (void)out_size; (void)ws_size;
  if (n_in < 132) return;
  const float* pr[132];
  for (int i = 0; i < 132; ++i) pr[i] = (const float*)d_in[i];
  const float* X = pr[0];
  const float* M0 = pr[1];

  int Dd[5] = {256, 128, 64, 32, 16};
  int Hh[5] = {256, 128, 64, 32, 16};
  int Ww[5] = {32, 16, 8, 4, 2};
  i64 nv[5];
  for (int l = 0; l < 5; ++l) nv[l] = (i64)Dd[l] * Hh[l] * Ww[l];

  char* ws = (char*)d_ws;
  size_t off = 0;
  auto alloc = [&](size_t bytes) -> void* {
    off = (off + 255) & ~(size_t)255;
    void* p = ws + off;
    off += bytes;
    return p;
  };

  const float* Mv[5];
  float* Mk[5];
  Mv[0] = M0;
  for (int l = 1; l < 5; ++l) { Mk[l] = (float*)alloc(nv[l] * 4); Mv[l] = Mk[l]; }
  f16* ENC[5];
  for (int l = 0; l < 5; ++l) ENC[l] = (f16*)alloc(nv[l] * 16 * 2);
  f16* T0 = (f16*)alloc(nv[0] * 16 * 2);
  f16* T1 = (f16*)alloc(nv[0] * 16 * 2);
  f16* T2 = (f16*)alloc(nv[0] * 16 * 2);
  f16* T3 = (f16*)alloc(nv[0] * 16 * 2);
  f16* Y  = (f16*)alloc(nv[0] * 16 * 2);
  f16* C32 = (f16*)alloc(nv[0] * 32 * 2);
  float* stats = (float*)alloc(65 * 4);
  float* sb = (float*)alloc(64 * 4);
  f16* PB = (f16*)alloc(32768);  // packed weight fragments (reused, stream-ordered)

  // geometry-templated dispatchers ------------------------------------------
  auto conv16 = [&](const f16* in, const float* w, f16* o, int l) {
    k_packw_pair<<<28, 256, 0, stream>>>(w, PB, 27, 14);
    unsigned nb = (unsigned)(nv[l] / 128);
    switch (l) {
      case 0: k_conv3<16, 256, 256, 32><<<nb, 256, 0, stream>>>(in, PB, o); break;
      case 1: k_conv3<16, 128, 128, 16><<<nb, 256, 0, stream>>>(in, PB, o); break;
      case 2: k_conv3<16, 64, 64, 8><<<nb, 256, 0, stream>>>(in, PB, o); break;
      case 3: k_conv3<16, 32, 32, 4><<<nb, 256, 0, stream>>>(in, PB, o); break;
      default: k_conv3<16, 16, 16, 2><<<nb, 256, 0, stream>>>(in, PB, o); break;
    }
  };
  auto conv32 = [&](const f16* in, const float* w, f16* o, int l) {
    k_packw32<<<54, 256, 0, stream>>>(w, PB);
    unsigned nb = (unsigned)(nv[l] / 128);
    switch (l) {
      case 0: k_conv3<32, 256, 256, 32><<<nb, 256, 0, stream>>>(in, PB, o); break;
      case 1: k_conv3<32, 128, 128, 16><<<nb, 256, 0, stream>>>(in, PB, o); break;
      case 2: k_conv3<32, 64, 64, 8><<<nb, 256, 0, stream>>>(in, PB, o); break;
      default: k_conv3<32, 32, 32, 4><<<nb, 256, 0, stream>>>(in, PB, o); break;
    }
  };
  auto down2 = [&](const f16* in, const float* w, f16* o, int l) { // l = out level 1..4
    k_packw_pair<<<8, 256, 0, stream>>>(w, PB, 8, 4);
    unsigned nb = (unsigned)(nv[l] / 128);
    switch (l) {
      case 1: k_down2<128, 128, 16><<<nb, 256, 0, stream>>>(in, PB, o); break;
      case 2: k_down2<64, 64, 8><<<nb, 256, 0, stream>>>(in, PB, o); break;
      case 3: k_down2<32, 32, 4><<<nb, 256, 0, stream>>>(in, PB, o); break;
      default: k_down2<16, 16, 2><<<nb, 256, 0, stream>>>(in, PB, o); break;
    }
  };
  auto up2 = [&](const f16* in, const float* w, f16* o, int lc) { // lc = coarse level 1..4
    k_packw_up<<<16, 256, 0, stream>>>(w, PB);
    unsigned nb = (unsigned)(nv[lc] / 16);
    switch (lc) {
      case 1: k_up2<128, 128, 16><<<nb, 256, 0, stream>>>(in, PB, o); break;
      case 2: k_up2<64, 64, 8><<<nb, 256, 0, stream>>>(in, PB, o); break;
      case 3: k_up2<32, 32, 4><<<nb, 256, 0, stream>>>(in, PB, o); break;
      default: k_up2<16, 16, 2><<<nb, 256, 0, stream>>>(in, PB, o); break;
    }
  };
  auto bnrun = [&](const f16* xin, const float* m, const float* g, const float* b,
                   int l, int C, int act, f16* o) {
    k_zero<<<1, 128, 0, stream>>>(stats, 65);
    int vpb = 256 / C;
    i64 nvx = nv[l];
    i64 sb64 = (nvx + vpb - 1) / vpb;
    unsigned sblocks = (unsigned)(sb64 > 4096 ? 4096 : sb64);
    k_bnstats<<<sblocks, 256, 0, stream>>>(xin, m, stats, nvx, C);
    k_bnfinal<<<1, 64, 0, stream>>>(stats, g, b, sb, C);
    i64 n = nvx * C;
    unsigned eb = (unsigned)((n + 255) / 256);
    k_bnapply<<<eb, 256, 0, stream>>>(xin, m, sb, o, nvx, C, act);
  };
  auto res16core = [&](const f16* x, int base, const float* m, int l, f16* tA, f16* tB) {
    bnrun(x, m, pr[base + 2], pr[base + 0], l, 16, 1, tA);
    conv16(tA, pr[base + 4], tB, l);
    bnrun(tB, m, pr[base + 3], pr[base + 1], l, 16, 1, tA);
    conv16(tA, pr[base + 5], tB, l);
  };
  auto resout16 = [&](const f16* x, const f16* cv, const float* m, int l, f16* o) {
    i64 n = nv[l] * 16;
    unsigned eb = (unsigned)((n + 255) / 256);
    k_resout<<<eb, 256, 0, stream>>>(x, cv, m, o, nv[l]);
  };

  // ---- mask pyramid ----
  for (int l = 1; l < 5; ++l) {
    unsigned eb = (unsigned)((nv[l] + 255) / 256);
    k_pool<<<eb, 256, 0, stream>>>(Mv[l - 1], Mk[l], Dd[l], Hh[l], Ww[l]);
  }

  // ---- stem ----
  const int ST = 62; // b1,b2,g1,g2,w1,w2
  {
    i64 n = nv[0] * 16;
    unsigned eb = (unsigned)((n + 255) / 256);
    k_mulmask<<<eb, 256, 0, stream>>>(X, Mv[0], T0, nv[0]);
  }
  conv16(T0, pr[ST + 4], T1, 0);
  bnrun(T1, Mv[0], pr[ST + 2], pr[ST + 0], 0, 16, 1, T0);
  conv16(T0, pr[ST + 5], T1, 0);
  bnrun(T1, Mv[0], pr[ST + 3], pr[ST + 1], 0, 16, 1, ENC[0]);

  // ---- encoder ----
  for (int i = 1; i <= 4; ++i) {
    int l = i;
    int dbase = 2 + (i - 1) * 3; // b,g,w
    down2(ENC[l - 1], pr[dbase + 2], T1, l);
    bnrun(T1, Mv[l], pr[dbase + 1], pr[dbase + 0], l, 16, 2, T0); // lrelu
    int r1 = 14 + ((i - 1) * 2 + 0) * 6;
    int r2 = 14 + ((i - 1) * 2 + 1) * 6;
    res16core(T0, r1, Mv[l], l, T1, T2);
    resout16(T0, T2, Mv[l], l, T3);
    res16core(T3, r2, Mv[l], l, T1, T2);
    resout16(T3, T2, Mv[l], l, ENC[i]);
  }

  // ---- decoder ----
  const f16* ycur = ENC[4];
  for (int i = 1; i <= 4; ++i) {
    int l = 4 - i, lc = l + 1;
    int ubase = 120 + (i - 1) * 3; // b,g,w
    up2(ycur, pr[ubase + 2], T1, lc);
    bnrun(T1, Mv[l], pr[ubase + 1], pr[ubase + 0], l, 16, 2, T0); // lrelu
    {
      i64 n = nv[l] * 32;
      unsigned eb = (unsigned)((n + 255) / 256);
      k_cat<<<eb, 256, 0, stream>>>(T0, ENC[l], C32, nv[l]);
    }
    int b1 = 68 + (i - 1) * 13; // u{i}r1: ba,bb,ga,gb,w1,w2,wd
    int b2 = b1 + 7;            // u{i}r2: ba,bb,ga,gb,w1,w2
    // residual block 32->16 with 1x1 downsample skip
    k_packw_d<<<2, 256, 0, stream>>>(pr[b1 + 6], PB);
    {
      unsigned nb = (unsigned)(nv[l] / 128);
      k_dense1x1<<<nb, 256, 0, stream>>>(C32, PB, T3);            // skip
    }
    bnrun(C32, Mv[l], pr[b1 + 2], pr[b1 + 0], l, 32, 1, C32);     // in-place bn+relu, 32ch
    conv32(C32, pr[b1 + 4], T1, l);
    bnrun(T1, Mv[l], pr[b1 + 3], pr[b1 + 1], l, 16, 1, T2);
    conv16(T2, pr[b1 + 5], T1, l);
    resout16(T3, T1, Mv[l], l, T0);
    // residual block 16->16
    res16core(T0, b2, Mv[l], l, T1, T2);
    if (i == 4) {
      i64 n = nv[0] * 16;
      unsigned eb = (unsigned)((n + 255) / 256);
      k_resout_f32<<<eb, 256, 0, stream>>>(T0, T2, Mv[0], (float*)d_out, nv[0]);
    } else {
      resout16(T0, T2, Mv[l], l, Y);
      ycur = Y;
    }
  }
}
